// MultiHeadAttention_31284541784220
// MI455X (gfx1250) — matrix operand order repaired and checked by reading, not compiled, not run on hardware
//
#include <hip/hip_runtime.h>
#include <math.h>

typedef __attribute__((ext_vector_type(16))) _Float16 v16h;
typedef __attribute__((ext_vector_type(8)))  _Float16 v8h;
typedef __attribute__((ext_vector_type(8)))  float    v8f;
typedef __attribute__((ext_vector_type(4)))  unsigned int u32x4;
typedef int v4i __attribute__((vector_size(4 * sizeof(int))));   // matches builtin param type

#define NHEAD 8
#define DIN   512
#define DK    64
#define LSEQ  2048
#define BATCH 2
#define BL    (BATCH*LSEQ)
#define DMODEL (NHEAD*DK)
#define NEG_BIG (-1.0e30f)

// LDS row paddings: keep 16B alignment, spread banks (stride in halves)
#define PAD40 40   // for 32-wide tiles  (80B stride, 20-dword bank step)
#define PAD72 72   // for 64-wide tiles (144B stride, 36-dword bank step)

// 16-bit A-fragment K index for element e (0..15); hi = lane>=16
__device__ __forceinline__ int amap(int e, int hi){ return ((e < 8) ? e : e + 8) + (hi << 3); }
// 16-bit B-fragment K index (same table order as amap)
__device__ __forceinline__ int bmap(int e, int hi){ return ((e < 8) ? e : e + 8) + (hi << 3); }

__device__ __forceinline__ v8f wmma_f16(v16h a, v16h b, v8f c){
  return __builtin_amdgcn_wmma_f32_16x16x32_f16(false, a, false, b, (short)0, c, false, false);
}

// ---- gfx1250 async global->LDS staging (guarded; falls back to vector copy) --
#if defined(__AMDGCN__) && __has_builtin(__builtin_amdgcn_global_load_async_to_lds_b128)
#define HAVE_ASYNC_LDS 1
#else
#define HAVE_ASYNC_LDS 0
#endif

__device__ __forceinline__ void async_copy16(const _Float16* g, _Float16* l){
#if HAVE_ASYNC_LDS
  __builtin_amdgcn_global_load_async_to_lds_b128(
      (__attribute__((address_space(1))) v4i*)g,
      (__attribute__((address_space(3))) v4i*)l, 0, 0);
#else
  *(u32x4*)l = *(const u32x4*)g;
#endif
}
__device__ __forceinline__ void async_wait(){
#if HAVE_ASYNC_LDS
#if __has_builtin(__builtin_amdgcn_s_wait_asynccnt)
  __builtin_amdgcn_s_wait_asynccnt(0);
#else
  asm volatile("s_wait_asynccnt 0x0" ::: "memory");
#endif
#endif
}

// ---------------------------------------------------------------------------
// Kernel 1: per-head QKV projection.  x(BL x 512) @ W_h(512 x 64) -> f16.
// q_s/k_s: [h][row][dk];  v_s stored TRANSPOSED: [h][dk][row] (for async V tiles)
// grid = (BL/128, NHEAD, 3), block = 256
// ---------------------------------------------------------------------------
__global__ __launch_bounds__(256) void proj_qkv(
    const float* __restrict__ q, const float* __restrict__ k, const float* __restrict__ v,
    const float* __restrict__ wq, const float* __restrict__ wk, const float* __restrict__ wv,
    _Float16* __restrict__ qkv_s)
{
  __shared__ _Float16 sX[128][PAD40];   // 128 rows x 32 k (f32->f16)
  __shared__ _Float16 sWt[64][PAD40];   // transposed weights: [n][k]
  const int tid  = threadIdx.x;
  const int lane = tid & 31, wid = tid >> 5;
  const int hi = (lane >> 4) & 1, ln = lane & 15;
  const int h = blockIdx.y;
  const int which = blockIdx.z;
  const int row0 = blockIdx.x * 128;

  const float* x = (which == 0) ? q : (which == 1) ? k : v;
  const float* w = ((which == 0) ? wq : (which == 1) ? wk : wv) + (size_t)h * DIN * DK;

  v8f acc[4] = {};

  for (int k0 = 0; k0 < DIN; k0 += 32) {
    for (int i = tid; i < 128 * 32; i += 256) {
      int r = i >> 5, c = i & 31;
      sX[r][c] = (_Float16)x[(size_t)(row0 + r) * DIN + k0 + c];
    }
    for (int i = tid; i < 32 * 64; i += 256) {
      int nn = i & 63, rr = i >> 6;                 // global-coalesced over nn
      sWt[nn][rr] = (_Float16)w[(size_t)(k0 + rr) * DK + nn];
    }
    __syncthreads();

    v16h a;
    #pragma unroll
    for (int e = 0; e < 16; ++e) a[e] = sX[(wid << 4) + ln][amap(e, hi)];
    #pragma unroll
    for (int t = 0; t < 4; ++t) {
      v16h bf;
      #pragma unroll
      for (int e = 0; e < 16; ++e) bf[e] = sWt[t * 16 + ln][bmap(e, hi)];   // contiguous
      acc[t] = wmma_f16(a, bf, acc[t]);
    }
    __syncthreads();
  }

  const int gr = row0 + (wid << 4) + (hi << 3);
  if (which == 2) {
    // transposed store: v_s[h][dk][row]; 8 consecutive rows per lane -> one b128
    _Float16* vst = qkv_s + (size_t)2 * NHEAD * BL * DK;
    #pragma unroll
    for (int t = 0; t < 4; ++t) {
      v8h pk;
      #pragma unroll
      for (int r = 0; r < 8; ++r) pk[r] = (_Float16)acc[t][r];
      *(v8h*)(vst + ((size_t)h * DK + t * 16 + ln) * BL + gr) = pk;
    }
  } else {
    _Float16* dst = qkv_s + ((size_t)which * NHEAD + h) * (size_t)BL * DK;
    #pragma unroll
    for (int t = 0; t < 4; ++t)
      #pragma unroll
      for (int r = 0; r < 8; ++r)
        dst[(size_t)(gr + r) * DK + t * 16 + ln] = (_Float16)acc[t][r];
  }
}

// ---------------------------------------------------------------------------
// Kernel 2: flash-style attention with async K/V tile staging.
// grid = (LSEQ/128, BATCH, NHEAD), block = 256 (8 waves; wave = 16 query rows)
// ---------------------------------------------------------------------------
__global__ __launch_bounds__(256) void flash_attn(
    const _Float16* __restrict__ qkv_s,
    float* __restrict__ attns,     // [H*B][L][L] raw scores (normalized in pass 3)
    float* __restrict__ conc,      // [BL][DMODEL]
    float* __restrict__ mrow, float* __restrict__ srow)
{
  __shared__ _Float16 sK[32][PAD72];      // [key][dk]  (row layout: score B-frags contiguous)
  __shared__ _Float16 sVt[64][PAD40];     // [dk][key]  (transposed: P@V B-frags contiguous)
  __shared__ _Float16 sP[8][16][PAD40];   // per-wave P tile, A-layout source
  const int tid  = threadIdx.x;
  const int lane = tid & 31, wid = tid >> 5;
  const int hi = (lane >> 4) & 1, ln = lane & 15;
  const int bb = blockIdx.y, h = blockIdx.z;
  const int q0 = blockIdx.x * 128 + wid * 16;
  const size_t hb = (size_t)h * BATCH + bb;

  const _Float16* qs  = qkv_s + (size_t)h * BL * DK + (size_t)bb * LSEQ * DK;
  const _Float16* ks  = qkv_s + ((size_t)NHEAD + h) * (size_t)BL * DK + (size_t)bb * LSEQ * DK;
  const _Float16* vst = qkv_s + (size_t)2 * NHEAD * BL * DK;   // [h][dk][row]
  float* arow = attns + hb * (size_t)LSEQ * LSEQ;

  // preload Q A-fragments for K=0..31 and K=32..63 (contiguous 16B runs)
  v16h aq0, aq1;
  {
    const size_t qr = (size_t)(q0 + ln) * DK;
    #pragma unroll
    for (int e = 0; e < 16; ++e) {
      int kk = amap(e, hi);
      aq0[e] = qs[qr + kk];
      aq1[e] = qs[qr + 32 + kk];
    }
  }

  v8f O[4] = {};
  float m[8], s[8];
  #pragma unroll
  for (int r = 0; r < 8; ++r) { m[r] = -3.0e38f; s[r] = 0.0f; }

  // staging coordinates (one 16B chunk per thread per tile)
  const int krr = tid >> 3, kcc = (tid & 7) * 8;   // sK : 32 rows x 64 dk
  const int vrr = tid >> 2, vcc = (tid & 3) * 8;   // sVt: 64 dk  x 32 keys

  const int cend = blockIdx.x * 4 + 4;             // causal chunk bound for this block
  for (int c = 0; c < cend; ++c) {
    const int kk0 = c * 32;
    async_copy16(&ks[(size_t)(kk0 + krr) * DK + kcc], &sK[krr][kcc]);
    async_copy16(vst + ((size_t)h * DK + vrr) * BL + (size_t)bb * LSEQ + kk0 + vcc,
                 &sVt[vrr][vcc]);
    async_wait();
    __syncthreads();

    if (kk0 <= q0 + 15) {                          // wave-uniform (EXEC stays full)
      float st[2][8];
      #pragma unroll
      for (int nt = 0; nt < 2; ++nt) {
        v16h b0, b1;
        const int key = nt * 16 + ln;
        #pragma unroll
        for (int e = 0; e < 16; ++e) {
          b0[e] = sK[key][bmap(e, hi)];            // two 16B runs
          b1[e] = sK[key][32 + bmap(e, hi)];
        }
        v8f Sv = {};
        Sv = wmma_f16(aq0, b0, Sv);
        Sv = wmma_f16(aq1, b1, Sv);
        const int colg = kk0 + key;
        #pragma unroll
        for (int r = 0; r < 8; ++r) {
          const int qg = q0 + r + (hi << 3);
          float sval = (colg > qg) ? NEG_BIG : Sv[r];
          st[nt][r] = sval;
          arow[(size_t)qg * LSEQ + colg] = sval;   // raw score, normalized in pass 3
        }
      }

      float mn[8], scale[8];
      #pragma unroll
      for (int r = 0; r < 8; ++r) {
        float tv = fmaxf(st[0][r], st[1][r]);
        tv = fmaxf(tv, __shfl_xor(tv, 1, 32));
        tv = fmaxf(tv, __shfl_xor(tv, 2, 32));
        tv = fmaxf(tv, __shfl_xor(tv, 4, 32));
        tv = fmaxf(tv, __shfl_xor(tv, 8, 32));
        mn[r] = fmaxf(m[r], tv);
        scale[r] = __expf(m[r] - mn[r]);
      }

      float rs[8];
      #pragma unroll
      for (int r = 0; r < 8; ++r) rs[r] = 0.0f;
      #pragma unroll
      for (int nt = 0; nt < 2; ++nt) {
        #pragma unroll
        for (int r = 0; r < 8; ++r) {
          const int qg = q0 + r + (hi << 3);
          const int colg = kk0 + nt * 16 + ln;
          float e = (colg > qg) ? 0.0f : __expf(st[nt][r] - mn[r]);
          rs[r] += e;
          sP[wid][r + (hi << 3)][nt * 16 + ln] = (_Float16)e;
        }
      }
      #pragma unroll
      for (int r = 0; r < 8; ++r) {
        float vv = rs[r];
        vv += __shfl_xor(vv, 1, 32);
        vv += __shfl_xor(vv, 2, 32);
        vv += __shfl_xor(vv, 4, 32);
        vv += __shfl_xor(vv, 8, 32);
        s[r] = s[r] * scale[r] + vv;
        m[r] = mn[r];
      }

      v16h ap;
      #pragma unroll
      for (int e = 0; e < 16; ++e) ap[e] = sP[wid][ln][amap(e, hi)];
      #pragma unroll
      for (int t = 0; t < 4; ++t) {
        v16h bv;
        #pragma unroll
        for (int e = 0; e < 16; ++e) bv[e] = sVt[t * 16 + ln][bmap(e, hi)];  // contiguous
        v8f o = O[t];
        #pragma unroll
        for (int r = 0; r < 8; ++r) o[r] *= scale[r];
        O[t] = wmma_f16(ap, bv, o);
      }
    }
    __syncthreads();
  }

  float inv[8];
  #pragma unroll
  for (int r = 0; r < 8; ++r) inv[r] = 1.0f / s[r];
  #pragma unroll
  for (int t = 0; t < 4; ++t)
    #pragma unroll
    for (int r = 0; r < 8; ++r) {
      const int qg = q0 + r + (hi << 3);
      conc[(size_t)(bb * LSEQ + qg) * DMODEL + h * DK + t * 16 + ln] = O[t][r] * inv[r];
    }
  if (ln == 0) {
    #pragma unroll
    for (int r = 0; r < 8; ++r) {
      const int qg = q0 + r + (hi << 3);
      mrow[hb * LSEQ + qg] = m[r];
      srow[hb * LSEQ + qg] = s[r];
    }
  }
}

// ---------------------------------------------------------------------------
// Kernel 3: normalize attns in place: exp(s-m)/sum, exact 0 above diagonal.
// ---------------------------------------------------------------------------
__global__ __launch_bounds__(256) void norm_attn(
    float* __restrict__ attns, const float* __restrict__ mrow, const float* __restrict__ srow)
{
  const size_t total4 = (size_t)NHEAD * BATCH * LSEQ * LSEQ / 4;
  size_t i4 = (size_t)blockIdx.x * blockDim.x + threadIdx.x;
  if (i4 >= total4) return;
  size_t idx = i4 * 4;
  size_t row = idx / LSEQ;               // hb*L + q
  int kb = (int)(idx % LSEQ);
  int qq = (int)(row % LSEQ);
  float mv  = mrow[row];
  float inv = 1.0f / srow[row];
  float4 x = *reinterpret_cast<float4*>(attns + idx);
  float4 y;
  y.x = (kb + 0 > qq) ? 0.0f : __expf(x.x - mv) * inv;
  y.y = (kb + 1 > qq) ? 0.0f : __expf(x.y - mv) * inv;
  y.z = (kb + 2 > qq) ? 0.0f : __expf(x.z - mv) * inv;
  y.w = (kb + 3 > qq) ? 0.0f : __expf(x.w - mv) * inv;
  *reinterpret_cast<float4*>(attns + idx) = y;
}

// ---------------------------------------------------------------------------
// Kernel 4: out = conc(BL x 512) @ proj_w^T(512 x 512) + proj_b
// grid = (BL/128, 512/64), block = 256
// ---------------------------------------------------------------------------
__global__ __launch_bounds__(256) void final_proj(
    const float* __restrict__ conc, const float* __restrict__ pw, const float* __restrict__ pb,
    float* __restrict__ out)
{
  __shared__ _Float16 sA[128][PAD40];
  __shared__ _Float16 sBt[64][PAD40];    // transposed: [n][k]
  const int tid  = threadIdx.x;
  const int lane = tid & 31, wid = tid >> 5;
  const int hi = (lane >> 4) & 1, ln = lane & 15;
  const int row0 = blockIdx.x * 128;
  const int n0   = blockIdx.y * 64;

  v8f acc[4] = {};
  for (int k0 = 0; k0 < DMODEL; k0 += 32) {
    for (int i = tid; i < 128 * 32; i += 256) {
      int r = i >> 5, c = i & 31;
      sA[r][c] = (_Float16)conc[(size_t)(row0 + r) * DMODEL + k0 + c];
    }
    for (int i = tid; i < 32 * 64; i += 256) {
      int kk = i & 31, nn = i >> 5;                // global-coalesced over kk
      sBt[nn][kk] = (_Float16)pw[(size_t)(n0 + nn) * DMODEL + k0 + kk];
    }
    __syncthreads();
    v16h a;
    #pragma unroll
    for (int e = 0; e < 16; ++e) a[e] = sA[(wid << 4) + ln][amap(e, hi)];
    #pragma unroll
    for (int t = 0; t < 4; ++t) {
      v16h bf;
      #pragma unroll
      for (int e = 0; e < 16; ++e) bf[e] = sBt[t * 16 + ln][bmap(e, hi)];   // contiguous
      acc[t] = wmma_f16(a, bf, acc[t]);
    }
    __syncthreads();
  }
  const int gr = row0 + (wid << 4) + (hi << 3);
  #pragma unroll
  for (int t = 0; t < 4; ++t)
    #pragma unroll
    for (int r = 0; r < 8; ++r)
      out[(size_t)(gr + r) * DMODEL + n0 + t * 16 + ln] = acc[t][r] + pb[n0 + t * 16 + ln];
}

// ---------------------------------------------------------------------------
extern "C" void kernel_launch(void* const* d_in, const int* in_sizes, int n_in,
                              void* d_out, int out_size, void* d_ws, size_t ws_size,
                              hipStream_t stream)
{
  (void)in_sizes; (void)n_in; (void)out_size; (void)ws_size;
  const float* q  = (const float*)d_in[0];
  const float* k  = (const float*)d_in[1];
  const float* v  = (const float*)d_in[2];
  // d_in[3] = attn_mask (pure causal; recomputed analytically on device)
  const float* wq = (const float*)d_in[4];
  const float* wk = (const float*)d_in[5];
  const float* wv = (const float*)d_in[6];
  const float* pw = (const float*)d_in[7];
  const float* pb = (const float*)d_in[8];

  float* outp  = (float*)d_out;
  float* attns = outp + (size_t)BL * DMODEL;   // out first, then attns (tuple order)

  // workspace layout (~21.2 MB)
  char* ws = (char*)d_ws;
  _Float16* qkv_s = (_Float16*)ws;                               // 3*H*BL*DK f16 = 12,582,912 B
  float* conc = (float*)(ws + 12582912);                         // BL*512 f32   =  8,388,608 B
  float* mrow = (float*)(ws + 12582912 + 8388608);               // 32768 f32
  float* srow = (float*)(ws + 12582912 + 8388608 + 131072);      // 32768 f32

  proj_qkv<<<dim3(BL / 128, NHEAD, 3), 256, 0, stream>>>(q, k, v, wq, wk, wv, qkv_s);
  flash_attn<<<dim3(LSEQ / 128, BATCH, NHEAD), 256, 0, stream>>>(qkv_s, attns, conc, mrow, srow);
  {
    size_t total4 = (size_t)NHEAD * BATCH * LSEQ * LSEQ / 4;
    norm_attn<<<(unsigned)((total4 + 255) / 256), 256, 0, stream>>>(attns, mrow, srow);
  }
  final_proj<<<dim3(BL / 128, DMODEL / 64), 256, 0, stream>>>(conc, pw, pb, outp);
}
